// VectorQuantizer_37632503448129
// MI455X (gfx1250) — matrix operand order repaired and checked by reading, not compiled, not run on hardware
//
#include <hip/hip_runtime.h>
#include <hip/hip_bf16.h>

// ---------------------------------------------------------------------------
// VQ codebook lookup for MI455X (gfx1250, wave32, WMMA).
// distances: argmin_j (||e_j||^2 - 2 * z.e_j)  -- ||z||^2 dropped (row const)
// GEMM in bf16 hi/lo split (3 x v_wmma_f32_16x16x32_bf16 per 16x16x32 tile)
// Round 3: grouped B-fragment loads (4 live fragments) so one s_wait_dscnt
// covers 6 WMMAs; double-buffered LDS staging with global prefetch retained.
// ---------------------------------------------------------------------------

typedef __bf16 bf16_t;
typedef __attribute__((ext_vector_type(16))) __bf16 v16bf;
typedef __attribute__((ext_vector_type(8)))  __bf16 v8bf;
typedef __attribute__((ext_vector_type(8)))  float  v8f;

#define C_DIM   256          // feature dim
#define N_ROWS  16384        // 16*32*32 z vectors
#define N_EMB   16384        // codebook entries
#define KSTEPS  8            // 256 / 32
#define WAVES   4            // waves per block in main kernel
#define MBLK    (WAVES * 16) // rows per block
#define NTILES  (N_EMB / 16)
#define LDSROW  (C_DIM + 8)  // bf16 elements per LDS row (pad vs bank conflicts)

// ---------------------------------------------------------------------------
// Pack fp32 rows into bf16 hi/lo pair + row L2-norm^2 (256 threads = 1 row)
// ---------------------------------------------------------------------------
__global__ void vq_pack_bf16_kernel(const float* __restrict__ src,
                                    bf16_t* __restrict__ hi,
                                    bf16_t* __restrict__ lo,
                                    float* __restrict__ norms) {
  const int row = blockIdx.x;
  const int t   = threadIdx.x;
  const float x = src[row * C_DIM + t];
  const bf16_t h = (bf16_t)x;
  const bf16_t l = (bf16_t)(x - (float)h);
  hi[row * C_DIM + t] = h;
  lo[row * C_DIM + t] = l;

  __shared__ float red[C_DIM];
  red[t] = x * x;
  __syncthreads();
  for (int s = 128; s > 0; s >>= 1) {
    if (t < s) red[t] += red[t + s];
    __syncthreads();
  }
  if (t == 0) norms[row] = red[0];
}

// ---------------------------------------------------------------------------
// helpers for the main kernel
// ---------------------------------------------------------------------------
struct StageRegs { v8bf h[4]; v8bf l[4]; };

__device__ __forceinline__ void stage_global(const bf16_t* __restrict__ ehi,
                                             const bf16_t* __restrict__ elo,
                                             int nbase, int tid, StageRegs& r) {
#pragma unroll
  for (int qi = 0; qi < 4; ++qi) {
    const int cc  = tid + qi * (WAVES * 32);  // 0..511 chunk id
    const int row = cc >> 5;                  // 0..15
    const int kc  = (cc & 31) << 3;           // 0..248 step 8
    r.h[qi] = *(const v8bf*)(&ehi[(size_t)(nbase + row) * C_DIM + kc]);
    r.l[qi] = *(const v8bf*)(&elo[(size_t)(nbase + row) * C_DIM + kc]);
  }
}

__device__ __forceinline__ void stage_store(bf16_t* __restrict__ sH,
                                            bf16_t* __restrict__ sL,
                                            int tid, const StageRegs& r) {
#pragma unroll
  for (int qi = 0; qi < 4; ++qi) {
    const int cc  = tid + qi * (WAVES * 32);
    const int row = cc >> 5;
    const int kc  = (cc & 31) << 3;
    *(v8bf*)(&sH[row * LDSROW + kc]) = r.h[qi];
    *(v8bf*)(&sL[row * LDSROW + kc]) = r.l[qi];
  }
}

__device__ __forceinline__ v16bf load_bfrag(const bf16_t* p) {
  v8bf b0 = *(const v8bf*)(p);
  v8bf b1 = *(const v8bf*)(p + 16);
  v16bf r;
#pragma unroll
  for (int e = 0; e < 8; ++e) { r[e] = b0[e]; r[8 + e] = b1[e]; }
  return r;
}

// ---------------------------------------------------------------------------
// Main GEMM + fused argmin. 4 waves / block; wave owns one 16-row M tile with
// full-K A fragments resident in registers; B tiles double-buffered in LDS;
// B fragments loaded two K-steps at a time (4 simultaneously-live fragments)
// so DS waits amortize over 6 back-to-back WMMAs.
// ---------------------------------------------------------------------------
__global__ __launch_bounds__(WAVES * 32, 1)
void vq_argmin_kernel(const bf16_t* __restrict__ zhi,
                      const bf16_t* __restrict__ zlo,
                      const bf16_t* __restrict__ ehi,
                      const bf16_t* __restrict__ elo,
                      const float*  __restrict__ enorm,
                      int*   __restrict__ idx_out,
                      float* __restrict__ idxf_out) {
  __shared__ bf16_t sH[2][16 * LDSROW];
  __shared__ bf16_t sL[2][16 * LDSROW];

  const int tid   = threadIdx.x;
  const int lane  = tid & 31;
  const int half  = lane >> 4;        // 0: lanes 0-15, 1: lanes 16-31
  const int lcol  = lane & 15;
  const int wave  = tid >> 5;
  const int mbase = blockIdx.x * MBLK + wave * 16;

  // ---- A fragments (z rows), full K=256, hi+lo, kept in VGPRs -------------
  // 16-bit A layout (16x32): lane%16 = M row; lanes0-15 hold K {0..7,16..23},
  // lanes16-31 hold K {8..15,24..31}  -> two contiguous 8-element chunks.
  const int arow  = mbase + lcol;
  const int koffA = half * 8;
  v16bf a_hi[KSTEPS], a_lo[KSTEPS];
#pragma unroll
  for (int ks = 0; ks < KSTEPS; ++ks) {
    const bf16_t* p = zhi + (size_t)arow * C_DIM + ks * 32 + koffA;
    v8bf p0 = *(const v8bf*)(p);
    v8bf p1 = *(const v8bf*)(p + 16);
    const bf16_t* q = zlo + (size_t)arow * C_DIM + ks * 32 + koffA;
    v8bf q0 = *(const v8bf*)(q);
    v8bf q1 = *(const v8bf*)(q + 16);
#pragma unroll
    for (int e = 0; e < 8; ++e) {
      a_hi[ks][e] = p0[e];  a_hi[ks][8 + e] = p1[e];
      a_lo[ks][e] = q0[e];  a_lo[ks][8 + e] = q1[e];
    }
  }

  float minval[8];
  int   minidx[8];
#pragma unroll
  for (int i = 0; i < 8; ++i) { minval[i] = 3.4e38f; minidx[i] = 0; }

  const int koffB = half * 8;  // B layout: lane%16 = N col, lane half h holds K {8h..8h+7, 16+8h..16+8h+7}

  // ---- prologue: stage tile 0 into LDS buffer 0 ---------------------------
  StageRegs stg;
  stage_global(ehi, elo, 0, tid, stg);
  stage_store(sH[0], sL[0], tid, stg);

  for (int nt = 0; nt < NTILES; ++nt) {
    const int cur   = nt & 1;
    const int nbase = nt * 16;
    const int ntn   = (nt + 1 < NTILES) ? nt + 1 : nt;  // clamp (uniform)

    __syncthreads();  // buf[cur] visible; all reads of buf[cur^1] retired

    // prefetch next tile global -> regs; latency hidden under 24 WMMAs
    stage_global(ehi, elo, ntn * 16, tid, stg);

    // ---- 16x16 tile of dot products: hi*hi + lo*hi + hi*lo ---------------
    const bf16_t* rowH = &sH[cur][lcol * LDSROW + koffB];
    const bf16_t* rowL = &sL[cur][lcol * LDSROW + koffB];
    v8f acc = {};
#pragma unroll
    for (int kg = 0; kg < KSTEPS / 2; ++kg) {
      const int k0 = 2 * kg;
      const int k1 = 2 * kg + 1;
      // 4 simultaneously-live B fragments: 8 ds_load_b128 issue together,
      // one dscnt wait covers the 6 WMMAs below.
      v16bf bh0 = load_bfrag(rowH + k0 * 32);
      v16bf bl0 = load_bfrag(rowL + k0 * 32);
      v16bf bh1 = load_bfrag(rowH + k1 * 32);
      v16bf bl1 = load_bfrag(rowL + k1 * 32);
      acc = __builtin_amdgcn_wmma_f32_16x16x32_bf16(
          false, a_hi[k0], false, bh0, (short)0, acc, false, false);
      acc = __builtin_amdgcn_wmma_f32_16x16x32_bf16(
          false, a_lo[k0], false, bh0, (short)0, acc, false, false);
      acc = __builtin_amdgcn_wmma_f32_16x16x32_bf16(
          false, a_hi[k0], false, bl0, (short)0, acc, false, false);
      acc = __builtin_amdgcn_wmma_f32_16x16x32_bf16(
          false, a_hi[k1], false, bh1, (short)0, acc, false, false);
      acc = __builtin_amdgcn_wmma_f32_16x16x32_bf16(
          false, a_lo[k1], false, bh1, (short)0, acc, false, false);
      acc = __builtin_amdgcn_wmma_f32_16x16x32_bf16(
          false, a_hi[k1], false, bl1, (short)0, acc, false, false);
    }

    // ---- fused argmin epilogue: C frag elem i = row (half*8+i), col lcol --
    const int   n  = nbase + lcol;
    const float en = enorm[n];
#pragma unroll
    for (int i = 0; i < 8; ++i) {
      const float s = en - 2.0f * acc[i];
      if (s < minval[i]) { minval[i] = s; minidx[i] = n; }
    }

    // ---- write prefetched tile into the other LDS buffer ------------------
    stage_store(sH[cur ^ 1], sL[cur ^ 1], tid, stg);
  }

  // ---- butterfly reduce over the 16 lanes sharing each row set ------------
#pragma unroll
  for (int m = 1; m < 16; m <<= 1) {
#pragma unroll
    for (int i = 0; i < 8; ++i) {
      const float ov = __shfl_xor(minval[i], m, 32);
      const int   oi = __shfl_xor(minidx[i], m, 32);
      if (ov < minval[i] || (ov == minval[i] && oi < minidx[i])) {
        minval[i] = ov; minidx[i] = oi;
      }
    }
  }
  if (lcol == 0) {
    const int rbase = mbase + half * 8;
#pragma unroll
    for (int i = 0; i < 8; ++i) {
      idx_out[rbase + i]  = minidx[i];
      idxf_out[rbase + i] = (float)minidx[i];
    }
  }
}

// ---------------------------------------------------------------------------
// z_q gather + per-row squared-diff sum (deterministic, no float atomics)
// ---------------------------------------------------------------------------
__global__ void vq_gather_kernel(const float* __restrict__ emb,
                                 const float* __restrict__ z,
                                 const int* __restrict__ idx,
                                 float* __restrict__ zq_out,
                                 float* __restrict__ row_sums) {
  const int row = blockIdx.x;
  const int t   = threadIdx.x;
  const int id  = idx[row];
  const float e = emb[(size_t)id * C_DIM + t];
  const float d = e - z[(size_t)row * C_DIM + t];
  zq_out[(size_t)row * C_DIM + t] = e;

  __shared__ float red[C_DIM];
  red[t] = d * d;
  __syncthreads();
  for (int s = 128; s > 0; s >>= 1) {
    if (t < s) red[t] += red[t + s];
    __syncthreads();
  }
  if (t == 0) row_sums[row] = red[0];
}

__global__ void vq_loss_kernel(const float* __restrict__ row_sums,
                               float* __restrict__ loss_out) {
  const int t = threadIdx.x;
  float s = 0.0f;
  for (int j = t; j < N_ROWS; j += 256) s += row_sums[j];
  __shared__ float red[256];
  red[t] = s;
  __syncthreads();
  for (int k = 128; k > 0; k >>= 1) {
    if (t < k) red[t] += red[t + k];
    __syncthreads();
  }
  // q_loss = BETA*mean((zq-z)^2) + mean((zq-z)^2) = 1.25*mean
  if (t == 0) loss_out[0] = 1.25f * red[0] / (float)((size_t)N_ROWS * C_DIM);
}

// ---------------------------------------------------------------------------
extern "C" void kernel_launch(void* const* d_in, const int* in_sizes, int n_in,
                              void* d_out, int out_size, void* d_ws, size_t ws_size,
                              hipStream_t stream) {
  (void)in_sizes; (void)n_in; (void)out_size; (void)ws_size;
  const float* z   = (const float*)d_in[0];  // (16,32,32,256) -> 16384 x 256
  const float* emb = (const float*)d_in[1];  // 16384 x 256

  // workspace layout
  char* ws = (char*)d_ws;
  const size_t ELEMS = (size_t)N_EMB * C_DIM;  // 4,194,304
  bf16_t* emb_hi = (bf16_t*)ws;
  bf16_t* emb_lo = emb_hi + ELEMS;
  bf16_t* z_hi   = emb_lo + ELEMS;
  bf16_t* z_lo   = z_hi + ELEMS;
  float*  e_norm = (float*)(z_lo + ELEMS);
  float*  z_norm = e_norm + N_EMB;   // computed but unused (argmin drops ||z||^2)
  int*    idx_i  = (int*)(z_norm + N_ROWS);
  float*  rsums  = (float*)(idx_i + N_ROWS);

  // output layout: z_q (4194304) | q_loss (1) | indices-as-float (16384)
  float* out    = (float*)d_out;
  float* zq_out = out;
  float* loss   = out + (size_t)N_ROWS * C_DIM;
  float* idxf   = loss + 1;

  vq_pack_bf16_kernel<<<N_ROWS, C_DIM, 0, stream>>>(z, z_hi, z_lo, z_norm);
  vq_pack_bf16_kernel<<<N_EMB, C_DIM, 0, stream>>>(emb, emb_hi, emb_lo, e_norm);

  vq_argmin_kernel<<<N_ROWS / MBLK, WAVES * 32, 0, stream>>>(
      z_hi, z_lo, emb_hi, emb_lo, e_norm, idx_i, idxf);

  vq_gather_kernel<<<N_ROWS, C_DIM, 0, stream>>>(emb, z, idx_i, zq_out, rsums);
  vq_loss_kernel<<<1, 256, 0, stream>>>(rsums, loss);
}
